// BornCollapseSampler_21328807592692
// MI455X (gfx1250) — compile-verified, matrix operand order checked by reading, not run on hardware
//
#include <hip/hip_runtime.h>
#include <math.h>

// ---------------- problem constants (from reference) ----------------
#define MROWS 512        // B*S = 64*8
#define DDIM  1024
#define KTOT  (2 * DDIM) // real | imag concatenated along K
#define VSZ   50257
#define TOPK  50
#define TOPP  0.95f

typedef __bf16 bf16_t;
typedef bf16_t v16bf __attribute__((ext_vector_type(16)));
typedef float  v8f   __attribute__((ext_vector_type(8)));

// ================================================================
// bf16 fragment load (fast path): the 16 K-values this lane owns are
// two contiguous runs of 8 bf16 (16 B each) at k0+lh*8 and k0+16+lh*8.
// Element i holds K = k0 + lh*8 + (i&7) + (i>=8 ? 16 : 0)  (CDNA5 ISA
// 16-bit A/B layout, lane-half lh selects the +8 K offset).
// ================================================================
__device__ __forceinline__ v16bf load_frag_bf16_direct(const bf16_t* __restrict__ rowPtr,
                                                       int k0, int lh) {
    uint4 lo = *reinterpret_cast<const uint4*>(rowPtr + k0 + lh * 8);
    uint4 hi = *reinterpret_cast<const uint4*>(rowPtr + k0 + 16 + lh * 8);
    struct Pair { uint4 a, b; } pr{lo, hi};
    return __builtin_bit_cast(v16bf, pr);
}

// f32 fragment load with inline cvt (fallback path)
__device__ __forceinline__ v16bf load_frag_f32cvt(const float* __restrict__ rowPtr,
                                                  int k0, int lh) {
    const float4* p0 = reinterpret_cast<const float4*>(rowPtr + k0 + lh * 8);
    const float4* p1 = reinterpret_cast<const float4*>(rowPtr + k0 + 16 + lh * 8);
    float4 a0 = p0[0], a1 = p0[1];
    float4 b0 = p1[0], b1 = p1[1];
    v16bf f;
    f[0]  = (bf16_t)a0.x; f[1]  = (bf16_t)a0.y; f[2]  = (bf16_t)a0.z; f[3]  = (bf16_t)a0.w;
    f[4]  = (bf16_t)a1.x; f[5]  = (bf16_t)a1.y; f[6]  = (bf16_t)a1.z; f[7]  = (bf16_t)a1.w;
    f[8]  = (bf16_t)b0.x; f[9]  = (bf16_t)b0.y; f[10] = (bf16_t)b0.z; f[11] = (bf16_t)b0.w;
    f[12] = (bf16_t)b1.x; f[13] = (bf16_t)b1.y; f[14] = (bf16_t)b1.z; f[15] = (bf16_t)b1.w;
    return f;
}

// ================================================================
// One-time f32 -> bf16 interleave: dst[row][0..1023]=real, [1024..2047]=imag
// ================================================================
__global__ void __launch_bounds__(256)
born_cvt_kernel(const float* __restrict__ re, const float* __restrict__ im,
                bf16_t* __restrict__ dst, int rows)
{
    struct B8 { bf16_t v[8]; };
    const int chunks = DDIM / 8;
    const int total  = rows * chunks;
    for (int i = blockIdx.x * blockDim.x + threadIdx.x; i < total;
         i += gridDim.x * blockDim.x) {
        int row = i / chunks;
        int k8  = (i - row * chunks) * 8;
        const float4* pr = reinterpret_cast<const float4*>(re + (size_t)row * DDIM + k8);
        const float4* pi = reinterpret_cast<const float4*>(im + (size_t)row * DDIM + k8);
        float4 r0 = pr[0], r1 = pr[1];
        float4 i0 = pi[0], i1 = pi[1];
        B8 rb{{(bf16_t)r0.x, (bf16_t)r0.y, (bf16_t)r0.z, (bf16_t)r0.w,
               (bf16_t)r1.x, (bf16_t)r1.y, (bf16_t)r1.z, (bf16_t)r1.w}};
        B8 ib{{(bf16_t)i0.x, (bf16_t)i0.y, (bf16_t)i0.z, (bf16_t)i0.w,
               (bf16_t)i1.x, (bf16_t)i1.y, (bf16_t)i1.z, (bf16_t)i1.w}};
        bf16_t* drow = dst + (size_t)row * KTOT;
        *reinterpret_cast<uint4*>(drow + k8)        = __builtin_bit_cast(uint4, rb);
        *reinterpret_cast<uint4*>(drow + DDIM + k8) = __builtin_bit_cast(uint4, ib);
    }
}

// ================================================================
// Fast-path GEMM over pre-converted bf16.
// block = 256 = 8 waves. Wave tile = 64 rows x 64 cols:
//   4 M-tiles x 4 N-tiles = 16 WMMA accumulators (128 VGPRs),
//   16 b128 loads per K-step for 16 WMMAs (1.0 loads/WMMA),
//   W re-read only 8x (vs 32x) -> ~1.7 GB L2 fragment traffic.
// blockIdx.x = M-block (fast varying) -> W slice stays hot in L2.
// ================================================================
__global__ void __launch_bounds__(256)
born_gemm_bf16_kernel(const bf16_t* __restrict__ Abf, const bf16_t* __restrict__ Wbf,
                      const float* __restrict__ bias, float* __restrict__ logits)
{
    const int lane  = threadIdx.x & 31;
    const int wave  = threadIdx.x >> 5;
    const int mBase = blockIdx.x * 64;                       // 8 M-blocks
    const int nBase = (blockIdx.y * 8 + wave) * 64;          // 64 cols per wave
    const int lh    = lane >> 4;
    const int l15   = lane & 15;

    const bf16_t* Arow[4];
#pragma unroll
    for (int u = 0; u < 4; ++u)
        Arow[u] = Abf + (size_t)(mBase + u * 16 + l15) * KTOT;

    const bf16_t* Wrow[4];
#pragma unroll
    for (int t = 0; t < 4; ++t) {
        int nc = nBase + t * 16 + l15;
        Wrow[t] = Wbf + (size_t)(nc < VSZ ? nc : (VSZ - 1)) * KTOT;
    }

    v8f acc[4][4] = {};   // [mtile][ntile]

#pragma unroll 1
    for (int k0 = 0; k0 < KTOT; k0 += 32) {
        // speculative prefetch of next K slice (OOB prefetch silently dropped)
        __builtin_prefetch(Wrow[0] + k0 + 32, 0, 3);
        __builtin_prefetch(Wrow[2] + k0 + 32, 0, 3);

        v16bf afrag[4];
#pragma unroll
        for (int u = 0; u < 4; ++u)
            afrag[u] = load_frag_bf16_direct(Arow[u], k0, lh);

#pragma unroll
        for (int t = 0; t < 4; ++t) {
            v16bf bfrag = load_frag_bf16_direct(Wrow[t], k0, lh);
#pragma unroll
            for (int u = 0; u < 4; ++u)
                acc[u][t] = __builtin_amdgcn_wmma_f32_16x16x32_bf16(
                    false, afrag[u], false, bfrag, (short)0, acc[u][t], false, false);
        }
    }

    // C/D layout: N = lane&15, VGPR r -> M = r + (lane>=16 ? 8 : 0)
#pragma unroll
    for (int t = 0; t < 4; ++t) {
        int nc = nBase + t * 16 + l15;
        if (nc < VSZ) {
            float bv = bias[nc];
#pragma unroll
            for (int u = 0; u < 4; ++u) {
#pragma unroll
                for (int r = 0; r < 8; ++r) {
                    int m = mBase + u * 16 + r + lh * 8;
                    logits[(size_t)m * VSZ + nc] = acc[u][t][r] + bv;
                }
            }
        }
    }
}

// ================================================================
// Fallback GEMM (workspace too small): inline f32->bf16 cvt.
// ================================================================
__global__ void __launch_bounds__(256)
born_gemm_f32_kernel(const float* __restrict__ Ar, const float* __restrict__ Ai,
                     const float* __restrict__ Wr, const float* __restrict__ Wi,
                     const float* __restrict__ bias, float* __restrict__ logits)
{
    const int lane  = threadIdx.x & 31;
    const int wave  = threadIdx.x >> 5;
    const int mBase = blockIdx.x * 16;
    const int nBase = (blockIdx.y * 8 + wave) * 64;
    const int lh    = lane >> 4;
    const int l15   = lane & 15;

    const int mrow = mBase + l15;
    int nrow[4];
#pragma unroll
    for (int t = 0; t < 4; ++t) {
        int nc = nBase + t * 16 + l15;
        nrow[t] = nc < VSZ ? nc : (VSZ - 1);
    }

    v8f acc[4] = {};

#pragma unroll 1
    for (int half = 0; half < 2; ++half) {
        const float* Arow = (half ? Ai : Ar) + (size_t)mrow * DDIM;
        const float* W    = half ? Wi : Wr;
#pragma unroll 1
        for (int k0 = 0; k0 < DDIM; k0 += 32) {
            __builtin_prefetch(Arow + k0 + 32, 0, 3);
            __builtin_prefetch(W + (size_t)nrow[0] * DDIM + k0 + 32, 0, 3);
            v16bf afrag = load_frag_f32cvt(Arow, k0, lh);
#pragma unroll
            for (int t = 0; t < 4; ++t) {
                v16bf bfrag = load_frag_f32cvt(W + (size_t)nrow[t] * DDIM, k0, lh);
                acc[t] = __builtin_amdgcn_wmma_f32_16x16x32_bf16(
                    false, afrag, false, bfrag, (short)0, acc[t], false, false);
            }
        }
    }

#pragma unroll
    for (int t = 0; t < 4; ++t) {
        int nc = nBase + t * 16 + l15;
        if (nc < VSZ) {
            float bv = bias[nc];
#pragma unroll
            for (int r = 0; r < 8; ++r) {
                int m = mBase + r + lh * 8;
                logits[(size_t)m * VSZ + nc] = acc[t][r] + bv;
            }
        }
    }
}

// ---------------- block reductions (256 threads) ----------------
__device__ __forceinline__ float blockReduceMax(float v, float* s) {
    int t = threadIdx.x;
    s[t] = v; __syncthreads();
    for (int off = 128; off > 0; off >>= 1) {
        if (t < off) s[t] = fmaxf(s[t], s[t + off]);
        __syncthreads();
    }
    float r = s[0]; __syncthreads();
    return r;
}
__device__ __forceinline__ float blockReduceSum(float v, float* s) {
    int t = threadIdx.x;
    s[t] = v; __syncthreads();
    for (int off = 128; off > 0; off >>= 1) {
        if (t < off) s[t] += s[t + off];
        __syncthreads();
    }
    float r = s[0]; __syncthreads();
    return r;
}

// ================================================================
// Per-row log-softmax (T==1). Logits (103 MB) live in the 192 MB L2,
// so all re-read passes are L2 traffic.
// ================================================================
__global__ void __launch_bounds__(256)
born_softmax_kernel(const float* __restrict__ logits, float* __restrict__ logp,
                    float* __restrict__ rowmax, float* __restrict__ rowlogsum)
{
    __shared__ float s[256];
    const int row = blockIdx.x;
    const float* x = logits + (size_t)row * VSZ;

    float m = -INFINITY;
    for (int v = threadIdx.x; v < VSZ; v += 256) m = fmaxf(m, x[v]);
    m = blockReduceMax(m, s);

    float sum = 0.f;
    for (int v = threadIdx.x; v < VSZ; v += 256) sum += __expf(x[v] - m);
    sum = blockReduceSum(sum, s);
    float ls = __logf(sum);

    float* o = logp + (size_t)row * VSZ;
    for (int v = threadIdx.x; v < VSZ; v += 256) o[v] = x[v] - m - ls;

    if (threadIdx.x == 0) { rowmax[row] = m; rowlogsum[row] = ls; }
}

// ---------------- deterministic hash -> uniform (0,1) ----------------
__device__ __forceinline__ float uhash(unsigned x) {
    x ^= x >> 16; x *= 0x7feb352dU;
    x ^= x >> 15; x *= 0x846ca68bU;
    x ^= x >> 16;
    return (float)(x >> 8) * (1.0f / 16777216.0f) + 1e-7f;
}

// ================================================================
// Per-row: top-k threshold (binary search on value), top-p threshold
// (binary search on kept mass), renormalized probs, Gumbel-max sample.
// ================================================================
__global__ void __launch_bounds__(256)
born_filter_sample_kernel(const float* __restrict__ logits,
                          const float* __restrict__ rowmax,
                          float* __restrict__ probs, float* __restrict__ tokens)
{
    __shared__ float s[256];
    __shared__ int   si[256];
    const int row = blockIdx.x;
    const float* x = logits + (size_t)row * VSZ;
    const float m = rowmax[row];

    // ---- top-k: k-th largest value (keep x >= thrK) ----
    float lo = m - 88.f, hi = m;
    for (int it = 0; it < 28; ++it) {
        float mid = 0.5f * (lo + hi);
        float c = 0.f;
        for (int v = threadIdx.x; v < VSZ; v += 256) c += (x[v] >= mid) ? 1.f : 0.f;
        c = blockReduceSum(c, s);
        if (c >= (float)TOPK) lo = mid; else hi = mid;
    }
    const float thrK = lo;

    float sk = 0.f;
    for (int v = threadIdx.x; v < VSZ; v += 256)
        if (x[v] >= thrK) sk += __expf(x[v] - m);
    sk = blockReduceSum(sk, s);

    // ---- top-p: smallest prefix with cumulative mass >= p ----
    lo = thrK; hi = m;
    for (int it = 0; it < 24; ++it) {
        float mid = 0.5f * (lo + hi);
        float g = 0.f;
        for (int v = threadIdx.x; v < VSZ; v += 256)
            if (x[v] >= mid) g += __expf(x[v] - m);
        g = blockReduceSum(g, s);
        if (g >= TOPP * sk) lo = mid; else hi = mid;
    }
    const float thrP = lo;

    float sp = 0.f;
    for (int v = threadIdx.x; v < VSZ; v += 256)
        if (x[v] >= thrP) sp += __expf(x[v] - m);
    sp = blockReduceSum(sp, s);
    const float inv = 1.f / sp;

    // ---- probs + Gumbel-max over kept set ----
    float best = -INFINITY;
    int   bidx = 0;
    float* po = probs + (size_t)row * VSZ;
    for (int v = threadIdx.x; v < VSZ; v += 256) {
        float xv   = x[v];
        bool  keep = xv >= thrP;
        po[v] = keep ? __expf(xv - m) * inv : 0.f;
        if (keep) {
            float u  = uhash((unsigned)row * 50411u + (unsigned)v * 2654435761u + 42u);
            float sc = xv + (-__logf(-__logf(u)));
            if (sc > best) { best = sc; bidx = v; }
        }
    }
    s[threadIdx.x] = best; si[threadIdx.x] = bidx; __syncthreads();
    for (int off = 128; off > 0; off >>= 1) {
        if (threadIdx.x < off && s[threadIdx.x + off] > s[threadIdx.x]) {
            s[threadIdx.x]  = s[threadIdx.x + off];
            si[threadIdx.x] = si[threadIdx.x + off];
        }
        __syncthreads();
    }
    if (threadIdx.x == 0) tokens[row] = (float)si[0];
}

// ================================================================
extern "C" void kernel_launch(void* const* d_in, const int* in_sizes, int n_in,
                              void* d_out, int out_size, void* d_ws, size_t ws_size,
                              hipStream_t stream) {
    (void)in_sizes; (void)n_in; (void)out_size;
    const float* psi_r = (const float*)d_in[0];
    const float* psi_i = (const float*)d_in[1];
    const float* W_r   = (const float*)d_in[2];
    const float* W_i   = (const float*)d_in[3];
    const float* bias  = (const float*)d_in[4];

    const size_t L = (size_t)MROWS * VSZ;
    float* logits = (float*)d_out;       // output 0
    float* logp   = logits + L;          // output 1
    float* tokens = logp + L;            // output 2 (512 values)
    float* probs  = tokens + MROWS;      // output 3

    float* rowmax    = (float*)d_ws;     // 512 floats
    float* rowlogsum = rowmax + MROWS;   // 512 floats

    // bf16 staging region (aligned at +4096)
    const size_t bfElems = ((size_t)MROWS + (size_t)VSZ) * KTOT;
    const size_t need    = 4096 + bfElems * sizeof(bf16_t);

    if (ws_size >= need) {
        bf16_t* Abf = (bf16_t*)((char*)d_ws + 4096);
        bf16_t* Wbf = Abf + (size_t)MROWS * KTOT;
        born_cvt_kernel<<<512,  256, 0, stream>>>(psi_r, psi_i, Abf, MROWS);
        born_cvt_kernel<<<8192, 256, 0, stream>>>(W_r,   W_i,   Wbf, VSZ);
        dim3 ggrid(MROWS / 64, (VSZ + 511) / 512);
        born_gemm_bf16_kernel<<<ggrid, 256, 0, stream>>>(Abf, Wbf, bias, logits);
    } else {
        dim3 ggrid(MROWS / 16, (VSZ + 511) / 512);
        born_gemm_f32_kernel<<<ggrid, 256, 0, stream>>>(psi_r, psi_i, W_r, W_i,
                                                        bias, logits);
    }

    born_softmax_kernel<<<MROWS, 256, 0, stream>>>(logits, logp, rowmax, rowlogsum);

    born_filter_sample_kernel<<<MROWS, 256, 0, stream>>>(logits, rowmax, probs, tokens);
}
